// WindowAttention3D_70025146794392
// MI455X (gfx1250) — compile-verified, matrix operand order checked by reading, not run on hardware
//
#include <hip/hip_runtime.h>
#include <hip/hip_bf16.h>

typedef __bf16 bf16_t;
typedef __attribute__((ext_vector_type(16))) __bf16 v16bf;
typedef __attribute__((ext_vector_type(8)))  __bf16 v8bf;
typedef __attribute__((ext_vector_type(8)))  float  v8f;
typedef __attribute__((ext_vector_type(4)))  float  v4f;
typedef __attribute__((ext_vector_type(4)))  unsigned int v4u;
typedef __attribute__((ext_vector_type(8)))  int    v8i;
typedef __attribute__((ext_vector_type(4)))  int    v4i;

#define B_WIN   2048
#define NTOK    98
#define NPAD    112      // 7 * 16
#define VPAD    128      // padded K for P*V
#define CDIM    256
#define NHEAD   8
#define DHEAD   32
#define NWIN    64
#define MROWS   (B_WIN * NTOK)          // 200704 (divisible by 128)
#define QK_SCALE 0.17677669529663687f   // 32^-0.5

#if __has_builtin(__builtin_amdgcn_tensor_load_to_lds) && \
    __has_builtin(__builtin_amdgcn_s_wait_tensorcnt)
#define USE_TDM 1
#else
#define USE_TDM 0
#endif

__device__ __forceinline__ v8f wmma_bf16(v16bf a, v16bf b, v8f c) {
  return __builtin_amdgcn_wmma_f32_16x16x32_bf16(false, a, false, b, (short)0, c,
                                                 false, false);
}

// A/B operand layout (ISA 7.12.2, 16-bit 16x32): lane holds K = 8*half+{0..7}
// in elements 0..7 and K = 8*half+16+{0..7} in elements 8..15.
__device__ __forceinline__ v16bf load16_bf(const bf16_t* p) {
  v8bf lo = *(const v8bf*)p;          // 16B load
  v8bf hi = *(const v8bf*)(p + 16);   // 16B load
  v16bf r;
#pragma unroll
  for (int i = 0; i < 8; ++i) { r[i] = lo[i]; r[i + 8] = hi[i]; }
  return r;
}

__device__ __forceinline__ v16bf load16_f32_cvt(const float* p) {
  const v4f* q = (const v4f*)p;
  v4f f0 = q[0], f1 = q[1], f2 = q[4], f3 = q[5];
  v16bf r;
#pragma unroll
  for (int i = 0; i < 4; ++i) {
    r[i]      = (bf16_t)f0[i];
    r[4 + i]  = (bf16_t)f1[i];
    r[8 + i]  = (bf16_t)f2[i];
    r[12 + i] = (bf16_t)f3[i];
  }
  return r;
}

// ---- Tensor Data Mover: stage a 2D bf16 tile (dim0 x dim1, row stride
// stride0 elements) from global into LDS. D# per ISA cdna5 ch.8.
__device__ __forceinline__ void tdm_load_2d(unsigned lds_off, const void* gptr,
                                            unsigned dim0, unsigned dim1,
                                            unsigned stride0) {
#if USE_TDM
  unsigned long long ga = (unsigned long long)(size_t)gptr;
  v4u g0;
  g0[0] = 1u;                                   // count=1, user mode
  g0[1] = lds_off;                              // lds_addr (bytes)
  g0[2] = (unsigned)ga;                         // global_addr[31:0]
  g0[3] = (unsigned)((ga >> 32) & 0x01FFFFFFu)  // global_addr[56:32]
          | 0x80000000u;                        // type=2 ("image")
  v8i g1;
  g1[0] = (int)(1u << 16);                      // data_size=1 -> 2 bytes
  g1[1] = (int)((dim0 & 0xFFFFu) << 16);        // tensor_dim0[15:0]  @bits 63:48
  g1[2] = (int)(((dim0 >> 16) & 0xFFFFu) |      // tensor_dim0[31:16]
                ((dim1 & 0xFFFFu) << 16));      // tensor_dim1[15:0]
  g1[3] = (int)(((dim1 >> 16) & 0xFFFFu) |      // tensor_dim1[31:16]
                ((dim0 & 0xFFFFu) << 16));      // tile_dim0 = dim0  @bits 127:112
  g1[4] = (int)(dim1 & 0xFFFFu);                // tile_dim1 = dim1; tile_dim2=0
  g1[5] = (int)stride0;                         // tensor_dim0_stride low 32
  g1[6] = 0;                                    // stride hi / dim1_stride
  g1[7] = 0;
  v4i gz4 = {0, 0, 0, 0};
  v8i gz8 = {0, 0, 0, 0, 0, 0, 0, 0};
  __builtin_amdgcn_tensor_load_to_lds(g0, g1, gz4, gz4, gz8, 0);
#else
  (void)lds_off; (void)gptr; (void)dim0; (void)dim1; (void)stride0;
#endif
}

__device__ __forceinline__ void tdm_wait() {
#if USE_TDM
  __builtin_amdgcn_s_wait_tensorcnt(0);
  asm volatile("" ::: "memory");
#endif
}

// fallback: cooperative 16B-chunk copy global -> LDS
__device__ __forceinline__ void coop_copy(bf16_t* dst, const bf16_t* src,
                                          int nchunks, int tid, int nthreads) {
  for (int i = tid; i < nchunks; i += nthreads)
    *(v8bf*)(dst + i * 8) = *(const v8bf*)(src + i * 8);
}

// ---- prep: transpose+cvt weights to bf16 column-major ----------------------
__global__ void prep_w(const float* __restrict__ qkv_w, const float* __restrict__ proj_w,
                       bf16_t* __restrict__ wT, bf16_t* __restrict__ pT) {
  int i = blockIdx.x * blockDim.x + threadIdx.x;
  if (i < 3 * CDIM * CDIM) {            // qkv_w [256][768] -> wT [768][256]
    int c = i / (3 * CDIM), d = i % (3 * CDIM);
    wT[(size_t)d * CDIM + c] = (bf16_t)qkv_w[i];
  }
  if (i < CDIM * CDIM) {                // proj_w [256][256] -> pT [256][256]
    int c = i >> 8, d = i & 255;
    pT[(size_t)d * CDIM + c] = (bf16_t)proj_w[i];
  }
}

// ---- prep: combined relative-position bias + window mask, padded -----------
__global__ void prep_bm(const float* __restrict__ mask, const float* __restrict__ bias_table,
                        float* __restrict__ bm) {
  int wh = blockIdx.x;                  // [0, 64*8)
  int w = wh >> 3, h = wh & 7;
  float* o = bm + (size_t)wh * NPAD * NPAD;
  const float* mk = mask + (size_t)w * NTOK * NTOK;
  for (int i = threadIdx.x; i < NPAD * NPAD; i += blockDim.x) {
    int n = i / NPAD, m = i % NPAD;     // n = query row, m = key col
    float v = -1.0e30f;                 // pad -> exp()==0 in softmax
    if (n < NTOK && m < NTOK) {
      int dn = n / 49, rn = n % 49, hn = rn / 7, wn = rn % 7;
      int dm = m / 49, rm = m % 49, hm = rm / 7, wm = rm % 7;
      int idx = (dn - dm + 1) * 169 + (hn - hm + 6) * 13 + (wn - wm + 6);
      v = bias_table[idx * NHEAD + h] + mk[n * NTOK + m];
    }
    o[i] = v;
  }
}

// ---- QKV GEMM: [200704,256] x [256,768], scatter to q/k/vT bf16 ------------
// grid = (12 col-tiles, 1568 row-tiles): col tiles of one row strip dispatch
// consecutively so the 131KB A-strip stays L2/L0 resident (x streams once).
__global__ void __launch_bounds__(256)
qkv_gemm(const float* __restrict__ x, const bf16_t* __restrict__ wT,
         bf16_t* __restrict__ q, bf16_t* __restrict__ k, bf16_t* __restrict__ vt) {
  __shared__ bf16_t wlds[64][CDIM];     // 32 KB staged B strip
  const int wid = threadIdx.x >> 5, lane = threadIdx.x & 31;
  const int half = lane >> 4, l16 = lane & 15;
  const int mbase = blockIdx.y * 128 + wid * 16;
  const int col0  = blockIdx.x * 64;

#if USE_TDM
  if (threadIdx.x < 32)
    tdm_load_2d((unsigned)(size_t)&wlds[0][0], wT + (size_t)col0 * CDIM,
                CDIM, 64, CDIM);
  tdm_wait();
#else
  coop_copy(&wlds[0][0], wT + (size_t)col0 * CDIM, 64 * CDIM / 8,
            threadIdx.x, 256);
#endif
  __syncthreads();

  v8f acc[4] = {};
  const float* xrow = x + (size_t)(mbase + l16) * CDIM;
#pragma unroll
  for (int it = 0; it < 8; ++it) {
    v16bf a = load16_f32_cvt(xrow + it * 32 + half * 8);
#pragma unroll
    for (int t = 0; t < 4; ++t) {
      v16bf b = load16_bf(&wlds[t * 16 + l16][it * 32 + half * 8]);
      acc[t] = wmma_bf16(a, b, acc[t]);
    }
  }
#pragma unroll
  for (int t = 0; t < 4; ++t) {
    const int d = col0 + t * 16 + l16;
    const int which = d >> 8, hd = (d >> 5) & 7, dh = d & 31;
#pragma unroll
    for (int r = 0; r < 8; ++r) {
      const int mm = mbase + r + half * 8;
      const int b = mm / NTOK, n = mm - b * NTOK;
      const float val = acc[t][r];
      if (which == 0)
        q[((size_t)(b * NHEAD + hd) * NTOK + n) * DHEAD + dh] = (bf16_t)(val * QK_SCALE);
      else if (which == 1)
        k[((size_t)(b * NHEAD + hd) * NTOK + n) * DHEAD + dh] = (bf16_t)val;
      else  // v stored transposed [B,H,32,VPAD] so P*V B-operand is contiguous
        vt[((size_t)(b * NHEAD + hd) * DHEAD + dh) * VPAD + n] = (bf16_t)val;
    }
  }
}

// ---- fused attention: S = qk^T (+bias+mask), softmax, O = P*v --------------
__global__ void __launch_bounds__(224)
win_attn(const bf16_t* __restrict__ q, const bf16_t* __restrict__ k,
         const bf16_t* __restrict__ vt, const float* __restrict__ bm,
         bf16_t* __restrict__ aout) {
  __shared__ bf16_t klds[NTOK][DHEAD];           // 6.25 KB: K tile (TDM staged)
  __shared__ bf16_t vlds[DHEAD][VPAD];           // 8 KB: V^T tile (TDM staged)
  __shared__ bf16_t plds[7][16][VPAD];           // 14 KB: per-wave P transpose
  const int bh = blockIdx.x, b = bh >> 3, h = bh & 7;
  const int wid = threadIdx.x >> 5, lane = threadIdx.x & 31;
  const int half = lane >> 4, l16 = lane & 15;

  const bf16_t* qb = q + (size_t)bh * NTOK * DHEAD;
  const bf16_t* kb = k + (size_t)bh * NTOK * DHEAD;
  const bf16_t* vb = vt + (size_t)bh * DHEAD * VPAD;
  const float*  bmb = bm + ((size_t)(b & (NWIN - 1)) * NHEAD + h) * NPAD * NPAD;

#if USE_TDM
  if (threadIdx.x < 32) {               // one TDM issue per WG (EXEC ignored)
    tdm_load_2d((unsigned)(size_t)&klds[0][0], kb, DHEAD, NTOK, DHEAD);
    tdm_load_2d((unsigned)(size_t)&vlds[0][0], vb, VPAD, DHEAD, VPAD);
  }
  tdm_wait();
#else
  coop_copy(&klds[0][0], kb, NTOK * DHEAD / 8, threadIdx.x, 224);
  coop_copy(&vlds[0][0], vb, DHEAD * VPAD / 8, threadIdx.x, 224);
#endif
  __syncthreads();

  // A = 16 query rows (clamped: rows >= 98 are garbage, never stored)
  const int mrow = wid * 16 + l16;
  const int mc = mrow < NTOK ? mrow : NTOK - 1;
  v16bf a = load16_bf(qb + (size_t)mc * DHEAD + half * 8);

  v8f s[7] = {};
#pragma unroll
  for (int t = 0; t < 7; ++t) {                  // S strip [16 x 112], K = Dh = 32
    const int nr = t * 16 + l16;
    const int nc = nr < NTOK ? nr : NTOK - 1;
    v16bf bk = load16_bf(&klds[nc][half * 8]);
    s[t] = wmma_bf16(a, bk, s[t]);
  }

  // bias+mask add and row softmax; a row lives in one 16-lane half-group
#pragma unroll
  for (int r = 0; r < 8; ++r) {
    const int row = wid * 16 + r + half * 8;
    const float* bmr = bmb + (size_t)row * NPAD;
    float mx = -3.0e38f;
#pragma unroll
    for (int t = 0; t < 7; ++t) {
      float v = s[t][r] + bmr[t * 16 + l16];
      s[t][r] = v;
      mx = fmaxf(mx, v);
    }
#pragma unroll
    for (int off = 1; off < 16; off <<= 1) mx = fmaxf(mx, __shfl_xor(mx, off, 32));
    float sum = 0.f;
#pragma unroll
    for (int t = 0; t < 7; ++t) {
      float e = __expf(s[t][r] - mx);
      s[t][r] = e;
      sum += e;
    }
#pragma unroll
    for (int off = 1; off < 16; off <<= 1) sum += __shfl_xor(sum, off, 32);
    const float inv = 1.f / sum;
#pragma unroll
    for (int t = 0; t < 7; ++t) s[t][r] *= inv;
  }

  // transpose P through LDS (C/D layout -> A layout); zero-pad cols [112,128)
#pragma unroll
  for (int r = 0; r < 8; ++r) {
    const int row = r + half * 8;
#pragma unroll
    for (int t = 0; t < 7; ++t) plds[wid][row][t * 16 + l16] = (bf16_t)s[t][r];
    plds[wid][row][NPAD + l16] = (bf16_t)0.f;
  }

  v8f o[2] = {};
#pragma unroll
  for (int u = 0; u < 4; ++u) {                  // K chunks of 32 over 128
    v16bf pa = load16_bf(&plds[wid][l16][u * 32 + half * 8]);
#pragma unroll
    for (int tc = 0; tc < 2; ++tc) {
      v16bf bv = load16_bf(&vlds[tc * 16 + l16][u * 32 + half * 8]);
      o[tc] = wmma_bf16(pa, bv, o[tc]);
    }
  }

#pragma unroll
  for (int tc = 0; tc < 2; ++tc)
#pragma unroll
    for (int r = 0; r < 8; ++r) {
      const int m = wid * 16 + r + half * 8;
      if (m < NTOK)
        aout[((size_t)b * NTOK + m) * CDIM + h * DHEAD + tc * 16 + l16] = (bf16_t)o[tc][r];
    }
}

// ---- output projection: [200704,256] x [256,256] + bias -> fp32 ------------
__global__ void __launch_bounds__(256)
proj_gemm(const bf16_t* __restrict__ aout, const bf16_t* __restrict__ pT,
          const float* __restrict__ proj_b, float* __restrict__ out) {
  __shared__ bf16_t wlds[64][CDIM];     // 32 KB staged B strip
  const int wid = threadIdx.x >> 5, lane = threadIdx.x & 31;
  const int half = lane >> 4, l16 = lane & 15;
  const int mbase = blockIdx.y * 128 + wid * 16;
  const int col0  = blockIdx.x * 64;

#if USE_TDM
  if (threadIdx.x < 32)
    tdm_load_2d((unsigned)(size_t)&wlds[0][0], pT + (size_t)col0 * CDIM,
                CDIM, 64, CDIM);
  tdm_wait();
#else
  coop_copy(&wlds[0][0], pT + (size_t)col0 * CDIM, 64 * CDIM / 8,
            threadIdx.x, 256);
#endif
  __syncthreads();

  v8f acc[4] = {};
  const bf16_t* arow = aout + (size_t)(mbase + l16) * CDIM;
#pragma unroll
  for (int it = 0; it < 8; ++it) {
    v16bf a = load16_bf(arow + it * 32 + half * 8);
#pragma unroll
    for (int t = 0; t < 4; ++t) {
      v16bf b = load16_bf(&wlds[t * 16 + l16][it * 32 + half * 8]);
      acc[t] = wmma_bf16(a, b, acc[t]);
    }
  }
#pragma unroll
  for (int t = 0; t < 4; ++t) {
    const int col = col0 + t * 16 + l16;
    const float pb = proj_b[col];
#pragma unroll
    for (int r = 0; r < 8; ++r) {
      const int mm = mbase + r + half * 8;
      out[(size_t)mm * CDIM + col] = acc[t][r] + pb;
    }
  }
}

extern "C" void kernel_launch(void* const* d_in, const int* in_sizes, int n_in,
                              void* d_out, int out_size, void* d_ws, size_t ws_size,
                              hipStream_t stream) {
  const float* x          = (const float*)d_in[0];
  const float* mask       = (const float*)d_in[1];
  const float* qkv_w      = (const float*)d_in[2];
  const float* proj_w     = (const float*)d_in[3];
  const float* proj_b     = (const float*)d_in[4];
  const float* bias_table = (const float*)d_in[5];
  float* out = (float*)d_out;
  (void)in_sizes; (void)n_in; (void)out_size; (void)ws_size;

  char* ws = (char*)d_ws;
  size_t off = 0;
  auto carve = [&](size_t bytes) -> void* {
    off = (off + 255) & ~(size_t)255;
    void* p = ws + off;
    off += bytes;
    return p;
  };
  bf16_t* qws  = (bf16_t*)carve((size_t)B_WIN * NHEAD * NTOK * DHEAD * sizeof(bf16_t));
  bf16_t* kws  = (bf16_t*)carve((size_t)B_WIN * NHEAD * NTOK * DHEAD * sizeof(bf16_t));
  bf16_t* vtws = (bf16_t*)carve((size_t)B_WIN * NHEAD * DHEAD * VPAD * sizeof(bf16_t));
  float*  bmws = (float*) carve((size_t)NWIN * NHEAD * NPAD * NPAD * sizeof(float));
  bf16_t* aout = (bf16_t*)carve((size_t)B_WIN * NTOK * CDIM * sizeof(bf16_t));
  bf16_t* wT   = (bf16_t*)carve((size_t)3 * CDIM * CDIM * sizeof(bf16_t));
  bf16_t* pT   = (bf16_t*)carve((size_t)CDIM * CDIM * sizeof(bf16_t));

  prep_w<<<(3 * CDIM * CDIM + 255) / 256, 256, 0, stream>>>(qkv_w, proj_w, wT, pT);
  prep_bm<<<NWIN * NHEAD, 256, 0, stream>>>(mask, bias_table, bmws);
  qkv_gemm<<<dim3(12, MROWS / 128), 256, 0, stream>>>(x, wT, qws, kws, vtws);
  win_attn<<<B_WIN * NHEAD, 224, 0, stream>>>(qws, kws, vtws, bmws, aout);
  proj_gemm<<<dim3(4, MROWS / 128), 256, 0, stream>>>(aout, pT, proj_b, out);
}
